// FusedLinearCrossEntropyLoss_25400436589206
// MI455X (gfx1250) — matrix operand order repaired and checked by reading, not compiled, not run on hardware
//
#include <hip/hip_runtime.h>
#include <hip/hip_bf16.h>

// ---------------------------------------------------------------------------
// Fused Linear + CrossEntropy for MI455X (gfx1250, wave32, WMMA)
//   hidden (N=32768, D=1024) f32, weight (V=4096, D=1024) f32, targets (N) int
// bf16 WMMA (16x16x32) w/ f32 accumulation; A block resident in LDS (hidden
// read once from HBM); weight streamed from L2 with double-buffered LDS
// staging; online softmax so the N x V logits tensor is never materialized.
// ---------------------------------------------------------------------------

typedef __attribute__((ext_vector_type(16))) __bf16        v16bf;
typedef __attribute__((ext_vector_type(8)))  float         v8f;
typedef __attribute__((ext_vector_type(4)))  float         v4f;
typedef __attribute__((ext_vector_type(4)))  unsigned int  v4u;
typedef __attribute__((ext_vector_type(2)))  unsigned int  v2u;

union Frag {
    v4u   u4[2];   // 8 dwords = 16 packed bf16
    v16bf b;
};

#define MBLK    128   // rows per workgroup (8 waves x 16 rows)
#define NTILE   64    // vocab columns per outer tile (4 x 16 per wave)
#define KCHUNK  64    // K staging chunk for B (2 WMMA K-steps)
#define THREADS 256

__device__ __forceinline__ unsigned int pack2_bf16(float lo, float hi) {
    unsigned int a = __float_as_uint(lo);
    unsigned int b = __float_as_uint(hi);
    a = (a + 0x7FFFu + ((a >> 16) & 1u)) >> 16;   // round-to-nearest-even
    b = (b + 0x7FFFu + ((b >> 16) & 1u)) >> 16;
    return a | (b << 16);
}

__global__ __launch_bounds__(THREADS)
void flce_main(const float* __restrict__ hidden,
               const float* __restrict__ weight,
               const int*   __restrict__ targets,
               float*       __restrict__ accbuf,   // [0]=sum(nll*mask) [1]=sum(mask)
               int N, int D, int V) {
    extern __shared__ unsigned short smem[];
    const int LDA = D + 8;         // padded bf16 row stride (bank-conflict free)
    const int LDB = KCHUNK + 8;
    unsigned short* Ash = smem;                          // MBLK x LDA
    unsigned short* Bsh = smem + MBLK * LDA;             // 2 x (NTILE x LDB)

    const int tid    = threadIdx.x;
    const int lane   = tid & 31;
    const int wave   = tid >> 5;
    const int hi     = lane >> 4;      // which 16-lane half
    const int ln     = lane & 15;
    const int wgRow0 = blockIdx.x * MBLK;

    // ---- stage A block: MBLK x D, fp32 -> bf16 into LDS (hidden read once) --
    {
        const int f4PerRow = D >> 2;
        const int total    = MBLK * f4PerRow;
        for (int idx = tid; idx < total; idx += THREADS) {
            const int row = idx / f4PerRow;
            const int c4  = idx - row * f4PerRow;
            const v4f h = *(const v4f*)(hidden + (size_t)(wgRow0 + row) * D + (c4 << 2));
            v2u p;
            p.x = pack2_bf16(h.x, h.y);
            p.y = pack2_bf16(h.z, h.w);
            *(v2u*)(Ash + row * LDA + (c4 << 2)) = p;
        }
    }

    // ---- loop-invariant staging descriptors for B chunks --------------------
    // NTILE*KCHUNK/4 float4 pieces == 4 * THREADS
    size_t wOff[4];
    int    bOff[4];
#pragma unroll
    for (int it = 0; it < 4; ++it) {
        const int idx = tid + it * THREADS;          // 0..1023
        const int n   = idx >> 4;                    // KCHUNK/4 == 16 float4 per row
        const int c4  = idx & 15;
        wOff[it] = (size_t)n * D + (c4 << 2);
        bOff[it] = n * LDB + (c4 << 2);
    }
    auto stageB = [&](int vbase, int kk, int bufIdx) {
        unsigned short* dst = Bsh + bufIdx * (NTILE * LDB);
        const size_t base = (size_t)vbase * D + kk;   // uniform scalar offset
#pragma unroll
        for (int it = 0; it < 4; ++it) {
            const v4f w = *(const v4f*)(weight + base + wOff[it]);
            v2u p;
            p.x = pack2_bf16(w.x, w.y);
            p.y = pack2_bf16(w.z, w.w);
            *(v2u*)(dst + bOff[it]) = p;
        }
        if (kk + 2 * KCHUNK <= D)                     // warm L2/L0 for chunk after next
            __builtin_prefetch(weight + base + (size_t)KCHUNK + wOff[tid & 3], 0, 1);
    };

    // ---- per-lane row state (each half-wave owns 8 rows of the 16-row tile) -
    const int rowBase = wgRow0 + wave * 16 + hi * 8;
    int tgt[8];
#pragma unroll
    for (int j = 0; j < 8; ++j) tgt[j] = targets[rowBase + j];

    float runmax[8], runsum[8], tl[8];
#pragma unroll
    for (int j = 0; j < 8; ++j) { runmax[j] = -3.0e38f; runsum[j] = 0.f; tl[j] = 0.f; }

    __syncthreads();   // A resident

    // ---- sweep the vocab in NTILE-column tiles -----------------------------
    for (int vbase = 0; vbase < V; vbase += NTILE) {
        v8f acc[4];
#pragma unroll
        for (int t = 0; t < 4; ++t)
#pragma unroll
            for (int j = 0; j < 8; ++j) acc[t][j] = 0.f;

        __syncthreads();            // previous tile's readers done
        stageB(vbase, 0, 0);        // prime buffer 0
        int buf = 0;

        for (int kk = 0; kk < D; kk += KCHUNK) {
            __syncthreads();        // staged chunk visible / safe to overwrite other buf
            if (kk + KCHUNK < D) stageB(vbase, kk + KCHUNK, buf ^ 1);

            // ---- load ALL fragments of this chunk first, then 8 WMMAs ------
            const unsigned short* Bcur = Bsh + buf * (NTILE * LDB);
            const unsigned short* ap   = Ash + (wave * 16 + ln) * LDA + kk + hi * 8;
            Frag a0, a1;
            a0.u4[0] = *(const v4u*)(ap);
            a0.u4[1] = *(const v4u*)(ap + 16);
            a1.u4[0] = *(const v4u*)(ap + 32);
            a1.u4[1] = *(const v4u*)(ap + 48);
            Frag b0[4], b1[4];
#pragma unroll
            for (int t = 0; t < 4; ++t) {
                const unsigned short* bp = Bcur + (t * 16 + ln) * LDB + hi * 8;
                b0[t].u4[0] = *(const v4u*)(bp);
                b0[t].u4[1] = *(const v4u*)(bp + 16);
                b1[t].u4[0] = *(const v4u*)(bp + 32);   // ks = 32 half
                b1[t].u4[1] = *(const v4u*)(bp + 48);
            }
#pragma unroll
            for (int t = 0; t < 4; ++t)
                acc[t] = __builtin_amdgcn_wmma_f32_16x16x32_bf16(
                    false, a0.b, false, b0[t].b, (short)0, acc[t], false, false);
#pragma unroll
            for (int t = 0; t < 4; ++t)
                acc[t] = __builtin_amdgcn_wmma_f32_16x16x32_bf16(
                    false, a1.b, false, b1[t].b, (short)0, acc[t], false, false);

            buf ^= 1;
        }

        // ---- online softmax update over this tile's 64 columns -------------
#pragma unroll
        for (int j = 0; j < 8; ++j) {
            const float c0 = acc[0][j], c1 = acc[1][j], c2 = acc[2][j], c3 = acc[3][j];
            const float mloc = fmaxf(fmaxf(c0, c1), fmaxf(c2, c3));
            const float nm   = fmaxf(runmax[j], mloc);
            const float s = __expf(c0 - nm) + __expf(c1 - nm) +
                            __expf(c2 - nm) + __expf(c3 - nm);
            runsum[j] = runsum[j] * __expf(runmax[j] - nm) + s;
            runmax[j] = nm;
#pragma unroll
            for (int t = 0; t < 4; ++t) {
                const int col = vbase + t * 16 + ln;
                if (col == tgt[j]) tl[j] += acc[t][j];   // exactly one lane matches
            }
        }
    }

    // ---- reduce across the 16 lanes holding each row's columns -------------
    float num = 0.f, den = 0.f;
#pragma unroll
    for (int j = 0; j < 8; ++j) {
        float m = runmax[j], s = runsum[j], t = tl[j];
#pragma unroll
        for (int off = 1; off < 16; off <<= 1) {
            const float om = __shfl_xor(m, off, 16);
            const float os = __shfl_xor(s, off, 16);
            const float nm = fmaxf(m, om);
            s = s * __expf(m - nm) + os * __expf(om - nm);
            m = nm;
            t += __shfl_xor(t, off, 16);
        }
        if (ln == 0 && tgt[j] != 0) {          // PAD_ID == 0
            num += m + __logf(s) - t;          // nll = max + log(sumexp) - logit_tgt
            den += 1.f;
        }
    }
    if (ln == 0) {
        unsafeAtomicAdd(&accbuf[0], num);      // global_atomic_add_f32
        unsafeAtomicAdd(&accbuf[1], den);
    }
}

__global__ void flce_final(const float* __restrict__ accbuf, float* __restrict__ out) {
    out[0] = accbuf[0] / accbuf[1];
}

extern "C" void kernel_launch(void* const* d_in, const int* in_sizes, int n_in,
                              void* d_out, int out_size, void* d_ws, size_t ws_size,
                              hipStream_t stream) {
    const float* hidden  = (const float*)d_in[0];
    const float* weight  = (const float*)d_in[1];
    const int*   targets = (const int*)d_in[2];
    float*       out     = (float*)d_out;
    float*       accbuf  = (float*)d_ws;

    const int Nrows = in_sizes[2];                 // B*T*NQ = 32768
    const int D     = in_sizes[0] / Nrows;         // 1024
    const int V     = in_sizes[1] / D;             // 4096

    hipMemsetAsync(accbuf, 0, 2 * sizeof(float), stream);

    const size_t smem = (size_t)(MBLK * (D + 8) + 2 * NTILE * (KCHUNK + 8)) * sizeof(unsigned short);
    dim3 grid(Nrows / MBLK);
    flce_main<<<grid, THREADS, smem, stream>>>(hidden, weight, targets, accbuf, Nrows, D, V);
    flce_final<<<1, 1, 0, stream>>>(accbuf, out);
}
